// NTXentLoss_5488968204621
// MI455X (gfx1250) — matrix operand order repaired and checked by reading, not compiled, not run on hardware
//
#include <hip/hip_runtime.h>

// ---------------------------------------------------------------------------
// NT-Xent loss, fused flash-softmax over E·E^T with bf16x3 WMMA (gfx1250).
//   E = concat(emb1, emb2) : 8192 x 512 f32 (L2-normalized rows)
//   out = mean_i [ logsumexp_{j!=i} (E_i·E_j / T) - E_i·E_lab(i) / T ]
// Each wave: 32 rows x one 4096-col half, 64-col groups, online softmax.
// Partial (m, s, z_label) per row merged by a small follow-up kernel.
// ---------------------------------------------------------------------------

typedef __attribute__((ext_vector_type(16))) __bf16 v16bf;
typedef __attribute__((ext_vector_type(8)))  float  v8f;

#define DIMK    512
#define NROWS   8192          // 2B
#define HALFR   4096          // B
#define KCHUNKS (DIMK / 32)   // 16
#define GPH     64            // 64-col groups per 4096-col half
#define WAVES   4             // waves per block

static constexpr float SCALE = 1.4426950408889634f / 0.07f;  // log2(e)/T
static constexpr float LN2   = 0.6931471805599453f;
static constexpr float NEG   = -3.0e38f;

// ---- conversion pass: f32 -> (bf16 hi, bf16 lo), concat emb1|emb2 ----------
__global__ void ntxent_convert(const float* __restrict__ e1,
                               const float* __restrict__ e2,
                               unsigned short* __restrict__ hi,
                               unsigned short* __restrict__ lo,
                               float* __restrict__ acc) {
    unsigned idx = blockIdx.x * blockDim.x + threadIdx.x;
    if (idx == 0) *acc = 0.0f;                 // reset loss accumulator
    if (idx >= (unsigned)(NROWS * DIMK)) return;
    float x = (idx < (unsigned)(HALFR * DIMK)) ? e1[idx] : e2[idx - HALFR * DIMK];
    unsigned u = __float_as_uint(x);           // rne bf16 split: x = hi + lo
    unsigned h = (u + 0x7FFFu + ((u >> 16) & 1u)) >> 16;
    float hf = __uint_as_float(h << 16);
    float l  = x - hf;
    unsigned ul = __float_as_uint(l);
    unsigned hl = (ul + 0x7FFFu + ((ul >> 16) & 1u)) >> 16;
    hi[idx] = (unsigned short)h;
    lo[idx] = (unsigned short)hl;
}

// ---- fragment loaders (ISA 7.12.2 element order, wave32) -------------------
// A 16x32 bf16: lanes 0-15 -> K {0..7,16..23}, lanes 16-31 -> K {8..15,24..31}
static __device__ __forceinline__ v16bf ldA(const unsigned short* p, int hiHalf) {
    union { v16bf v; uint4 u[2]; } f;
    const unsigned short* q = p + (hiHalf ? 8 : 0);
    f.u[0] = *reinterpret_cast<const uint4*>(q);
    f.u[1] = *reinterpret_cast<const uint4*>(q + 16);
    return f.v;
}
// B 32x16 bf16: lanes 0-15 -> K {0..7,16..23}, lanes 16-31 -> K {8..15,24..31}
static __device__ __forceinline__ v16bf ldB(const unsigned short* p, int hiHalf) {
    union { v16bf v; uint4 u[2]; } f;
    const unsigned short* q = p + (hiHalf ? 8 : 0);
    f.u[0] = *reinterpret_cast<const uint4*>(q);
    f.u[1] = *reinterpret_cast<const uint4*>(q + 16);
    return f.v;
}

static __device__ __forceinline__ v8f wmma_bf16(v16bf a, v16bf b, v8f c) {
    return __builtin_amdgcn_wmma_f32_16x16x32_bf16(
        false, a, false, b, (short)0, c, false, false);
}

// ---- main fused kernel: one wave = 32 rows x 4096 cols ---------------------
__global__ __launch_bounds__(32 * WAVES)
void ntxent_main(const unsigned short* __restrict__ eh,
                 const unsigned short* __restrict__ el,
                 float* __restrict__ ws_m,     // [2][8192] partial max
                 float* __restrict__ ws_s,     // [2][8192] partial sum 2^x
                 float* __restrict__ ws_zl) {  // [2][8192] label logit (or NEG)
    const int lane  = threadIdx.x & 31;
    const int wave  = threadIdx.x >> 5;
    const int wid   = blockIdx.x * WAVES + wave;   // 0 .. 511
    const int h     = wid & 1;                     // column half
    const int row_base = (wid >> 1) * 32;          // 32-row block
    const int lsel  = lane >> 4;                   // lane half
    const int nIdx  = lane & 15;                   // N (and A-row) lane index
    const int hi8   = lsel * 8;                    // accumulator M offset
    const int cg0   = h * GPH;                     // first 64-col group

    const unsigned short* arow_h[2];
    const unsigned short* arow_l[2];
#pragma unroll
    for (int t = 0; t < 2; ++t) {
        arow_h[t] = eh + (row_base + t * 16 + nIdx) * DIMK;
        arow_l[t] = el + (row_base + t * 16 + nIdx) * DIMK;
    }

    float m[2][8], s[2][8], zl[2][8];
#pragma unroll
    for (int t = 0; t < 2; ++t)
#pragma unroll
        for (int r = 0; r < 8; ++r) { m[t][r] = NEG; s[t][r] = 0.0f; zl[t][r] = NEG; }

    for (int cg = cg0; cg < cg0 + GPH; ++cg) {
        v8f acc8[2][4];
#pragma unroll
        for (int t = 0; t < 2; ++t)
#pragma unroll
            for (int g = 0; g < 4; ++g) acc8[t][g] = (v8f){0,0,0,0,0,0,0,0};

        const int cbase = cg * 64;
        for (int k = 0; k < KCHUNKS; ++k) {
            const int ko = k * 32;
            v16bf aH[2], aL[2];
#pragma unroll
            for (int t = 0; t < 2; ++t) {
                aH[t] = ldA(arow_h[t] + ko, lsel);
                aL[t] = ldA(arow_l[t] + ko, lsel);
            }
#pragma unroll
            for (int g = 0; g < 4; ++g) {
                const int col = cbase + g * 16 + nIdx;
                v16bf bH = ldB(eh + col * DIMK + ko, lsel);
                v16bf bL = ldB(el + col * DIMK + ko, lsel);
#pragma unroll
                for (int t = 0; t < 2; ++t) {
                    acc8[t][g] = wmma_bf16(aH[t], bH, acc8[t][g]);   // hi*hi
                    acc8[t][g] = wmma_bf16(aH[t], bL, acc8[t][g]);   // hi*lo
                    acc8[t][g] = wmma_bf16(aL[t], bH, acc8[t][g]);   // lo*hi
                }
            }
        }

        // ---- online softmax epilogue ----
        // All acc8 indices are compile-time constants (g fully unrolled);
        // label/diag selection uses uniform compares + per-lane cndmask,
        // so no dynamic register indexing -> no scratch spill.
#pragma unroll
        for (int t = 0; t < 2; ++t) {
            const int rb = row_base + t * 16;
            const int lb = rb + ((rb < HALFR) ? HALFR : -HALFR);
            const bool hasD = (rb >> 6) == cg;     // diagonal tile in group
            const bool hasL = (lb >> 6) == cg;     // label tile in group
            const int  gd   = (rb >> 4) & 3;
            const int  gl   = (lb >> 4) & 3;
#pragma unroll
            for (int r = 0; r < 8; ++r) {
                const int  M        = r + hi8;
                const bool selfElem = (nIdx == M);  // element (M, N=M%16)
                float xmax = NEG;
#pragma unroll
                for (int g = 0; g < 4; ++g) {
                    float x = acc8[t][g][r] * SCALE;
                    if (hasL && (g == gl) && selfElem) zl[t][r] = x;  // label
                    if (hasD && (g == gd) && selfElem) x = NEG;       // diag
                    xmax = fmaxf(xmax, x);
                }
                if (xmax > m[t][r]) {
                    s[t][r] *= __builtin_amdgcn_exp2f(m[t][r] - xmax);
                    m[t][r] = xmax;
                }
#pragma unroll
                for (int g = 0; g < 4; ++g) {
                    float x = acc8[t][g][r] * SCALE;
                    if (hasD && (g == gd) && selfElem) x = NEG;       // -> 0
                    s[t][r] += __builtin_amdgcn_exp2f(x - m[t][r]);
                }
            }
        }
    }

    // ---- merge 16 lanes of each half (shfl_xor stays within half) ----
#pragma unroll
    for (int t = 0; t < 2; ++t)
#pragma unroll
        for (int r = 0; r < 8; ++r) {
            for (int off = 1; off <= 8; off <<= 1) {
                float mo = __shfl_xor(m[t][r], off, 32);
                float so = __shfl_xor(s[t][r], off, 32);
                float mn = fmaxf(m[t][r], mo);
                s[t][r] = s[t][r] * __builtin_amdgcn_exp2f(m[t][r] - mn)
                        + so      * __builtin_amdgcn_exp2f(mo      - mn);
                m[t][r] = mn;
                zl[t][r] = fmaxf(zl[t][r], __shfl_xor(zl[t][r], off, 32));
            }
        }

    if (nIdx == 0) {   // lane 0 -> rows M 0-7, lane 16 -> rows M 8-15
#pragma unroll
        for (int t = 0; t < 2; ++t)
#pragma unroll
            for (int r = 0; r < 8; ++r) {
                const int row = row_base + t * 16 + hi8 + r;
                const int o   = h * NROWS + row;
                ws_m[o]  = m[t][r];
                ws_s[o]  = s[t][r];
                ws_zl[o] = zl[t][r];
            }
    }
}

// ---- merge the two column-half partials per row, accumulate mean ----------
__global__ void ntxent_merge(const float* __restrict__ ws_m,
                             const float* __restrict__ ws_s,
                             const float* __restrict__ ws_zl,
                             float* __restrict__ acc) {
    unsigned i = blockIdx.x * blockDim.x + threadIdx.x;
    if (i >= (unsigned)NROWS) return;
    float m0 = ws_m[i],          m1 = ws_m[NROWS + i];
    float s0 = ws_s[i],          s1 = ws_s[NROWS + i];
    float z  = fmaxf(ws_zl[i], ws_zl[NROWS + i]);   // exactly one is real
    float mt = fmaxf(m0, m1);
    float st = s0 * __builtin_amdgcn_exp2f(m0 - mt)
             + s1 * __builtin_amdgcn_exp2f(m1 - mt);
    float nll = LN2 * (mt + __builtin_amdgcn_logf(st) - z);
    atomicAdd(acc, nll);
}

// ---- finalize: mean --------------------------------------------------------
__global__ void ntxent_finalize(const float* __restrict__ acc,
                                float* __restrict__ out) {
    if (threadIdx.x == 0) out[0] = acc[0] * (1.0f / (float)NROWS);
}

extern "C" void kernel_launch(void* const* d_in, const int* in_sizes, int n_in,
                              void* d_out, int out_size, void* d_ws, size_t ws_size,
                              hipStream_t stream) {
    const float* emb1 = (const float*)d_in[0];
    const float* emb2 = (const float*)d_in[1];
    float* out = (float*)d_out;

    const size_t PLANE = (size_t)NROWS * DIMK;
    unsigned short* wsH = (unsigned short*)d_ws;          // 8 MB bf16 hi
    unsigned short* wsL = wsH + PLANE;                    // 8 MB bf16 lo
    float* ws_m  = (float*)(wsL + PLANE);                 // 2*8192 f32
    float* ws_s  = ws_m + 2 * NROWS;
    float* ws_zl = ws_s + 2 * NROWS;
    float* acc   = ws_zl + 2 * NROWS;                     // 1 f32

    {   // pass 1: concat + bf16 hi/lo split + zero accumulator
        const int total = NROWS * DIMK;
        ntxent_convert<<<(total + 255) / 256, 256, 0, stream>>>(
            emb1, emb2, wsH, wsL, acc);
    }
    {   // pass 2: fused GEMM + flash log-softmax partials (512 waves)
        const int blocks = (NROWS / 32) * 2 / WAVES;      // 128
        ntxent_main<<<blocks, 32 * WAVES, 0, stream>>>(
            wsH, wsL, ws_m, ws_s, ws_zl);
    }
    ntxent_merge<<<(NROWS + 255) / 256, 256, 0, stream>>>(ws_m, ws_s, ws_zl, acc);
    ntxent_finalize<<<1, 32, 0, stream>>>(acc, out);
}
